// CodecTransformer_16406775070873
// MI455X (gfx1250) — matrix operand order repaired and checked by reading, not compiled, not run on hardware
//
#include <hip/hip_runtime.h>

typedef __attribute__((ext_vector_type(16))) __bf16 v16bf;
typedef __attribute__((ext_vector_type(8)))  __bf16 v8bf;
typedef __attribute__((ext_vector_type(8)))  float  v8f;

#define B_  2
#define T_  2048
#define C_  2048
#define H_  16
#define KV_ 4
#define HD_ 128
#define BT_ (B_ * T_)          // 4096

// ---------- helpers ----------

__device__ __forceinline__ unsigned short f2bf(float f) {
  unsigned u = __float_as_uint(f);
  unsigned r = u + 0x7fffu + ((u >> 16) & 1u);   // round-to-nearest-even
  return (unsigned short)(r >> 16);
}

__device__ __forceinline__ v8f wmma_bf16(v16bf a, v16bf b, v8f c) {
  // D = A(16x32) * B(32x16) + C, fp32 accumulate
  return __builtin_amdgcn_wmma_f32_16x16x32_bf16(false, a, false, b, (short)0, c,
                                                 false, false);
}

__device__ __forceinline__ v16bf mk16(v8bf lo, v8bf hi) {
  v16bf r;
#pragma unroll
  for (int i = 0; i < 8; ++i) { r[i] = lo[i]; r[8 + i] = hi[i]; }
  return r;
}

// A fragment from a per-lane pointer (lane offset m*ld + g*8 already applied);
// element offset e is a compile-time constant -> immediate load offsets.
// A 16x32 layout: lane m=lane&15, g=lane>>4; K = g*8+h (v0-3), 16+g*8+h (v4-7)
__device__ __forceinline__ v16bf frag_a(const __bf16* p, int e) {
  return mk16(*(const v8bf*)(p + e), *(const v8bf*)(p + e + 16));
}

// B fragment from a per-lane pointer (lane offset n*ld + g*8 already applied).
// B 32x16 layout: lane n=lane&15, g=lane>>4; K = g*8+h (v0-3), 16+g*8+h (v4-7)
__device__ __forceinline__ v16bf frag_b(const __bf16* p, int e) {
  return mk16(*(const v8bf*)(p + e), *(const v8bf*)(p + e + 16));
}

// Generic loaders (used where the base changes every call)
__device__ __forceinline__ v16bf load_frag_a(const __bf16* base, int ld) {
  int lane = threadIdx.x & 31;
  const __bf16* p = base + (size_t)(lane & 15) * ld + (lane >> 4) * 8;
  return frag_a(p, 0);
}
__device__ __forceinline__ v16bf load_frag_b(const __bf16* baseT, int ld) {
  int lane = threadIdx.x & 31;
  const __bf16* p = baseT + (size_t)(lane & 15) * ld + (lane >> 4) * 8;
  return frag_b(p, 0);
}

// ---------- conversion kernels ----------

__global__ void f32_to_bf16_kernel(const float* __restrict__ src,
                                   unsigned short* __restrict__ dst, int n) {
  int i = blockIdx.x * blockDim.x + threadIdx.x;
  if (i < n) dst[i] = f2bf(src[i]);
}

// W[k][n] (K rows, N cols) -> Wt[n][k] bf16
__global__ void transpose_f32_to_bf16_kernel(const float* __restrict__ W,
                                             unsigned short* __restrict__ Wt,
                                             int K, int N) {
  int i = blockIdx.x * blockDim.x + threadIdx.x;
  if (i < K * N) {
    int k = i / N, n = i - k * N;
    Wt[(size_t)n * K + k] = f2bf(W[i]);
  }
}

// ---------- generic bf16 GEMM: C[M][N] = A[M][K] * Bt[n][k]^T ----------
// block = 256 threads = 8 waves; each wave computes a 32(M) x 64(N) tile.
// K-loop unrolled by 2 with explicit stage-A/stage-B fragment sets so the
// software pipeline needs no register copies; per-lane base pointers advance
// by 64 elements per double-step and loads use immediate offsets.
// Requires K % 64 == 0 and K >= 128. grid = (N/64, M/256)

__global__ __launch_bounds__(256) void gemm_bf16_kernel(
    const unsigned short* __restrict__ A, const unsigned short* __restrict__ Bt,
    float* __restrict__ C, int M, int N, int K) {
  int wave = threadIdx.x >> 5;
  int lane = threadIdx.x & 31;
  int idx = lane & 15, g = lane >> 4;
  int m0 = blockIdx.y * 256 + wave * 32;
  int n0 = blockIdx.x * 64;

  // per-lane steady pointers
  const __bf16* pa0 = (const __bf16*)A + ((size_t)(m0 + idx) * K) + g * 8;
  const __bf16* pa1 = pa0 + (size_t)16 * K;
  const __bf16* pb0 = (const __bf16*)Bt + ((size_t)(n0 + idx) * K) + g * 8;
  const __bf16* pb1 = pb0 + (size_t)16 * K;
  const __bf16* pb2 = pb0 + (size_t)32 * K;
  const __bf16* pb3 = pb0 + (size_t)48 * K;

  v8f acc0[4], acc1[4];
#pragma unroll
  for (int nt = 0; nt < 4; ++nt)
#pragma unroll
    for (int i = 0; i < 8; ++i) { acc0[nt][i] = 0.f; acc1[nt][i] = 0.f; }

  // prologue: stage X at k = 0
  v16bf xA0 = frag_a(pa0, 0), xA1 = frag_a(pa1, 0);
  v16bf xB0 = frag_b(pb0, 0), xB1 = frag_b(pb1, 0);
  v16bf xB2 = frag_b(pb2, 0), xB3 = frag_b(pb3, 0);

  for (int k0 = 0; k0 + 64 < K; k0 += 64) {
    // stage Y loads (k0 + 32)
    v16bf yA0 = frag_a(pa0, 32), yA1 = frag_a(pa1, 32);
    v16bf yB0 = frag_b(pb0, 32), yB1 = frag_b(pb1, 32);
    v16bf yB2 = frag_b(pb2, 32), yB3 = frag_b(pb3, 32);
    __builtin_prefetch(pa0 + 64, 0, 1);
    __builtin_prefetch(pb0 + 64, 0, 1);

    // compute stage X (k0)
    acc0[0] = wmma_bf16(xA0, xB0, acc0[0]);
    acc1[0] = wmma_bf16(xA1, xB0, acc1[0]);
    acc0[1] = wmma_bf16(xA0, xB1, acc0[1]);
    acc1[1] = wmma_bf16(xA1, xB1, acc1[1]);
    acc0[2] = wmma_bf16(xA0, xB2, acc0[2]);
    acc1[2] = wmma_bf16(xA1, xB2, acc1[2]);
    acc0[3] = wmma_bf16(xA0, xB3, acc0[3]);
    acc1[3] = wmma_bf16(xA1, xB3, acc1[3]);

    // advance and load stage X for k0 + 64
    pa0 += 64; pa1 += 64; pb0 += 64; pb1 += 64; pb2 += 64; pb3 += 64;
    xA0 = frag_a(pa0, 0); xA1 = frag_a(pa1, 0);
    xB0 = frag_b(pb0, 0); xB1 = frag_b(pb1, 0);
    xB2 = frag_b(pb2, 0); xB3 = frag_b(pb3, 0);

    // compute stage Y (k0 + 32)
    acc0[0] = wmma_bf16(yA0, yB0, acc0[0]);
    acc1[0] = wmma_bf16(yA1, yB0, acc1[0]);
    acc0[1] = wmma_bf16(yA0, yB1, acc0[1]);
    acc1[1] = wmma_bf16(yA1, yB1, acc1[1]);
    acc0[2] = wmma_bf16(yA0, yB2, acc0[2]);
    acc1[2] = wmma_bf16(yA1, yB2, acc1[2]);
    acc0[3] = wmma_bf16(yA0, yB3, acc0[3]);
    acc1[3] = wmma_bf16(yA1, yB3, acc1[3]);
  }

  // tail: k0 = K-64 (stage X loaded), stage Y = K-32
  {
    v16bf yA0 = frag_a(pa0, 32), yA1 = frag_a(pa1, 32);
    v16bf yB0 = frag_b(pb0, 32), yB1 = frag_b(pb1, 32);
    v16bf yB2 = frag_b(pb2, 32), yB3 = frag_b(pb3, 32);

    acc0[0] = wmma_bf16(xA0, xB0, acc0[0]);
    acc1[0] = wmma_bf16(xA1, xB0, acc1[0]);
    acc0[1] = wmma_bf16(xA0, xB1, acc0[1]);
    acc1[1] = wmma_bf16(xA1, xB1, acc1[1]);
    acc0[2] = wmma_bf16(xA0, xB2, acc0[2]);
    acc1[2] = wmma_bf16(xA1, xB2, acc1[2]);
    acc0[3] = wmma_bf16(xA0, xB3, acc0[3]);
    acc1[3] = wmma_bf16(xA1, xB3, acc1[3]);

    acc0[0] = wmma_bf16(yA0, yB0, acc0[0]);
    acc1[0] = wmma_bf16(yA1, yB0, acc1[0]);
    acc0[1] = wmma_bf16(yA0, yB1, acc0[1]);
    acc1[1] = wmma_bf16(yA1, yB1, acc1[1]);
    acc0[2] = wmma_bf16(yA0, yB2, acc0[2]);
    acc1[2] = wmma_bf16(yA1, yB2, acc1[2]);
    acc0[3] = wmma_bf16(yA0, yB3, acc0[3]);
    acc1[3] = wmma_bf16(yA1, yB3, acc1[3]);
  }

#pragma unroll
  for (int nt = 0; nt < 4; ++nt)
#pragma unroll
    for (int i = 0; i < 8; ++i) {
      C[(size_t)(m0 + i + 8 * g) * N + n0 + nt * 16 + idx] = acc0[nt][i];
      C[(size_t)(m0 + 16 + i + 8 * g) * N + n0 + nt * 16 + idx] = acc1[nt][i];
    }
}

// ---------- RoPE / packing ----------

__device__ __forceinline__ float rope_val(const float* __restrict__ row,
                                          int col, int d, int t) {
  int j = d & 63;
  float inv = __expf(-(float)j * 0.14391156831f);  // ln(10000)/64
  float s, c;
  __sincosf((float)t * inv, &s, &c);
  float x0 = row[col];
  float xp = (d < 64) ? -row[col + 64] : row[col - 64];
  return x0 * c + xp * s;
}

// Qf[4096][2048] -> Qb bf16 [b][h][t][d]
__global__ void rope_q_kernel(const float* __restrict__ Qf,
                              unsigned short* __restrict__ Qb) {
  int i = blockIdx.x * blockDim.x + threadIdx.x;   // < 4096*2048
  int row = i >> 11, col = i & 2047;
  int h = col >> 7, d = col & 127;
  int b = row >> 11, t = row & 2047;
  float v = rope_val(Qf + (size_t)row * 2048, col, d, t);
  Qb[(((size_t)(b * H_ + h)) * T_ + t) * HD_ + d] = f2bf(v);
}

// Kf[4096][512] -> kout fp32 [b][kvh][t][d]  and  Kb bf16 same layout
__global__ void rope_k_kernel(const float* __restrict__ Kf,
                              float* __restrict__ kout,
                              unsigned short* __restrict__ Kb) {
  int i = blockIdx.x * blockDim.x + threadIdx.x;   // < 4096*512
  int row = i >> 9, col = i & 511;
  int kvh = col >> 7, d = col & 127;
  int b = row >> 11, t = row & 2047;
  float v = rope_val(Kf + (size_t)row * 512, col, d, t);
  size_t o = (((size_t)(b * KV_ + kvh)) * T_ + t) * HD_ + d;
  kout[o] = v;
  Kb[o] = f2bf(v);
}

// Vf[4096][512] -> vout fp32 [b][kvh][t][d]  and  Vbt bf16 [b][kvh][d][t]
__global__ void pack_v_kernel(const float* __restrict__ Vf,
                              float* __restrict__ vout,
                              unsigned short* __restrict__ Vbt) {
  int i = blockIdx.x * blockDim.x + threadIdx.x;   // < 4096*512
  int row = i >> 9, col = i & 511;
  int kvh = col >> 7, d = col & 127;
  int b = row >> 11, t = row & 2047;
  float v = Vf[i];
  vout[(((size_t)(b * KV_ + kvh)) * T_ + t) * HD_ + d] = v;
  Vbt[(((size_t)(b * KV_ + kvh)) * HD_ + d) * T_ + t] = f2bf(v);
}

// ---------- flash attention (causal, GQA) ----------
// One wave handles one 16-query tile of one head. 8 waves / block.
// S = Q*K^T (4 WMMA per 16-key subtile), online softmax, P staged through LDS
// (C/D-layout -> A-layout transpose), O += P*V (8 WMMA per 32-key block).
// V fragments are fetched right after the S-WMMAs so the loads overlap with
// the softmax VALU work and the LDS round trip.

__global__ __launch_bounds__(256) void attn_kernel(
    const unsigned short* __restrict__ Qb, const unsigned short* __restrict__ Kb,
    const unsigned short* __restrict__ Vbt, unsigned short* __restrict__ Ob) {
  __shared__ unsigned short plds[8 * 16 * 32];   // 8 KB: one 16x32 P tile per wave

  int wave = threadIdx.x >> 5;
  int lane = threadIdx.x & 31;
  int idx = lane & 15, g = lane >> 4;

  int tile = blockIdx.x * 8 + wave;      // 4096 tiles total
  int head = tile >> 7;                  // 128 query tiles per head
  int t0 = (tile & 127) << 4;
  int b = head >> 4, h = head & 15, kvh = h >> 2;

  const __bf16* Qh = (const __bf16*)Qb + (size_t)(b * H_ + h) * T_ * HD_;
  const __bf16* Kh = (const __bf16*)Kb + (size_t)(b * KV_ + kvh) * T_ * HD_;
  const __bf16* Vh = (const __bf16*)Vbt + (size_t)(b * KV_ + kvh) * HD_ * T_;
  unsigned short* pw = plds + wave * 512;

  v16bf qf[4];
#pragma unroll
  for (int dd = 0; dd < 4; ++dd)
    qf[dd] = load_frag_a(Qh + (size_t)t0 * HD_ + dd * 32, HD_);

  float m_i[8], l_i[8];
  v8f oacc[8];
#pragma unroll
  for (int i = 0; i < 8; ++i) { m_i[i] = -1e30f; l_i[i] = 0.f; }
#pragma unroll
  for (int nt = 0; nt < 8; ++nt)
#pragma unroll
    for (int i = 0; i < 8; ++i) oacc[nt][i] = 0.f;

  const float scale = 0.08838834764831845f;  // 1/sqrt(128)
  int smax = t0 + 15;

  for (int s0 = 0; s0 <= smax; s0 += 32) {
    v8f st[2];
#pragma unroll
    for (int hf = 0; hf < 2; ++hf) {
#pragma unroll
      for (int i = 0; i < 8; ++i) st[hf][i] = 0.f;
      int kb0 = s0 + hf * 16;
      if (kb0 <= smax) {
#pragma unroll
        for (int dd = 0; dd < 4; ++dd) {
          v16bf kf = load_frag_b(Kh + (size_t)kb0 * HD_ + dd * 32, HD_);
          st[hf] = wmma_bf16(qf[dd], kf, st[hf]);
        }
      }
    }

    // issue V-fragment loads now; consumed after the softmax + LDS round trip
    v16bf vf[8];
#pragma unroll
    for (int nt = 0; nt < 8; ++nt)
      vf[nt] = load_frag_b(Vh + (size_t)(nt * 16) * T_ + s0, T_);

    // online softmax per owned row r = i + 8*g; cols striped over 16-lane halves
#pragma unroll
    for (int i = 0; i < 8; ++i) {
      int r = i + 8 * g;
      int trow = t0 + r;
      bool v0 = (s0 + idx) <= trow;
      bool v1 = (s0 + 16 + idx) <= trow;
      float a0 = v0 ? st[0][i] * scale : -1e30f;
      float a1 = v1 ? st[1][i] * scale : -1e30f;
      float lm = fmaxf(a0, a1);
#pragma unroll
      for (int msk = 1; msk < 16; msk <<= 1)
        lm = fmaxf(lm, __shfl_xor(lm, msk, 32));
      float mnew = fmaxf(m_i[i], lm);
      float alpha = __expf(m_i[i] - mnew);
      float p0 = v0 ? __expf(a0 - mnew) : 0.f;
      float p1 = v1 ? __expf(a1 - mnew) : 0.f;
      float rs = p0 + p1;
#pragma unroll
      for (int msk = 1; msk < 16; msk <<= 1) rs += __shfl_xor(rs, msk, 32);
      l_i[i] = l_i[i] * alpha + rs;
      m_i[i] = mnew;
#pragma unroll
      for (int nt = 0; nt < 8; ++nt) oacc[nt][i] *= alpha;
      pw[r * 32 + idx] = f2bf(p0);        // row-major 16x32 P tile
      pw[r * 32 + 16 + idx] = f2bf(p1);
    }

    // P (A operand) back from LDS; O += P * V   (same-wave LDS ops are in-order)
    v16bf pf = load_frag_a((const __bf16*)pw, 32);
#pragma unroll
    for (int nt = 0; nt < 8; ++nt)
      oacc[nt] = wmma_bf16(pf, vf[nt], oacc[nt]);
  }

  // epilogue: O / l, store bf16 as [b][t][h][d] (== [B*T, 2048] row-major)
#pragma unroll
  for (int i = 0; i < 8; ++i) {
    float inv_l = 1.0f / l_i[i];
    int r = t0 + i + 8 * g;
#pragma unroll
    for (int nt = 0; nt < 8; ++nt) {
      Ob[(((size_t)(b * T_ + r)) * H_ + h) * HD_ + nt * 16 + idx] =
          f2bf(oacc[nt][i] * inv_l);
    }
  }
}

// ---------- launch ----------

extern "C" void kernel_launch(void* const* d_in, const int* in_sizes, int n_in,
                              void* d_out, int out_size, void* d_ws, size_t ws_size,
                              hipStream_t stream) {
  (void)in_sizes; (void)n_in; (void)out_size; (void)ws_size;
  const float* x  = (const float*)d_in[0];
  const float* Wq = (const float*)d_in[1];
  const float* Wk = (const float*)d_in[2];
  const float* Wv = (const float*)d_in[3];
  const float* Wo = (const float*)d_in[4];
  float* out = (float*)d_out;
  float* kout = out + (size_t)BT_ * C_;                       // 8,388,608
  float* vout = kout + (size_t)B_ * KV_ * T_ * HD_;           // +2,097,152

  size_t off = 0;
  auto alloc = [&](size_t bytes) {
    void* p = (char*)d_ws + off;
    off += (bytes + 255) & ~(size_t)255;
    return p;
  };
  unsigned short* xb   = (unsigned short*)alloc((size_t)BT_ * C_ * 2);      // 16 MB
  unsigned short* wqT  = (unsigned short*)alloc((size_t)2048 * 2048 * 2);   // 8 MB
  unsigned short* wkT  = (unsigned short*)alloc((size_t)512 * 2048 * 2);    // 2 MB
  unsigned short* wvT  = (unsigned short*)alloc((size_t)512 * 2048 * 2);    // 2 MB
  unsigned short* woT  = (unsigned short*)alloc((size_t)2048 * 2048 * 2);   // 8 MB
  float*          projf= (float*)alloc((size_t)BT_ * 2048 * 4);             // 32 MB (Q/K/V reuse)
  unsigned short* qb   = (unsigned short*)alloc((size_t)BT_ * 2048 * 2);    // 16 MB
  unsigned short* kb   = (unsigned short*)alloc((size_t)B_ * KV_ * T_ * HD_ * 2); // 4 MB
  unsigned short* vbt  = (unsigned short*)alloc((size_t)B_ * KV_ * T_ * HD_ * 2); // 4 MB
  unsigned short* ob   = (unsigned short*)alloc((size_t)BT_ * 2048 * 2);    // 16 MB

  const int TB = 256;
  // 1) convert inputs
  f32_to_bf16_kernel<<<(BT_ * C_) / TB, TB, 0, stream>>>(x, xb, BT_ * C_);
  transpose_f32_to_bf16_kernel<<<(2048 * 2048) / TB, TB, 0, stream>>>(Wq, wqT, 2048, 2048);
  transpose_f32_to_bf16_kernel<<<(2048 * 512) / TB, TB, 0, stream>>>(Wk, wkT, 2048, 512);
  transpose_f32_to_bf16_kernel<<<(2048 * 512) / TB, TB, 0, stream>>>(Wv, wvT, 2048, 512);
  transpose_f32_to_bf16_kernel<<<(2048 * 2048) / TB, TB, 0, stream>>>(Wo, woT, 2048, 2048);

  // 2) Q projection + RoPE
  gemm_bf16_kernel<<<dim3(2048 / 64, BT_ / 256), TB, 0, stream>>>(xb, wqT, projf, BT_, 2048, 2048);
  rope_q_kernel<<<(BT_ * 2048) / TB, TB, 0, stream>>>(projf, qb);

  // 3) K projection + RoPE (+ fp32 output)
  gemm_bf16_kernel<<<dim3(512 / 64, BT_ / 256), TB, 0, stream>>>(xb, wkT, projf, BT_, 512, 2048);
  rope_k_kernel<<<(BT_ * 512) / TB, TB, 0, stream>>>(projf, kout, kb);

  // 4) V projection (+ fp32 output, transposed bf16 for attention)
  gemm_bf16_kernel<<<dim3(512 / 64, BT_ / 256), TB, 0, stream>>>(xb, wvT, projf, BT_, 512, 2048);
  pack_v_kernel<<<(BT_ * 512) / TB, TB, 0, stream>>>(projf, vout, vbt);

  // 5) causal GQA flash attention: 4096 wave-tiles, 8 per block
  attn_kernel<<<512, TB, 0, stream>>>(qb, kb, vbt, ob);

  // 6) output projection
  gemm_bf16_kernel<<<dim3(2048 / 64, BT_ / 256), TB, 0, stream>>>(ob, woT, out, BT_, 2048, 2048);
}
